// AdditiveSoftAttention_67946382623344
// MI455X (gfx1250) — matrix operand order repaired and checked by reading, not compiled, not run on hardware
//
#include <hip/hip_runtime.h>
#include <hip/hip_bf16.h>
#include <math.h>

#define B_   64
#define S_   2048
#define RNN_ 1024
#define ATT_ 512
#define CTX_ 1024
#define MIN_VALUE_ (-100000000.0f)

typedef __attribute__((ext_vector_type(16))) _Float16 v16h;
typedef __attribute__((ext_vector_type(8)))  float    v8f;

// TDM descriptor vector types (6-arg clang-23 builtin form:
//   (uint32x4 g0, int32x8 g1, int32x4 g2, int32x4 g3, int32x8 g4, i32 cpol))
typedef __attribute__((ext_vector_type(4))) unsigned int tdm_v4u;
typedef __attribute__((ext_vector_type(8))) int          tdm_v8i;
typedef __attribute__((ext_vector_type(4))) int          tdm_v4i;

// ---------------------------------------------------------------------------
// Issue a TDM load of a 2D tile [tile_y rows x tile_x elts] of f32 from
// global (row stride row_stride_elts) into LDS at byte offset lds_off.
// D# packing per CDNA5 ISA ch.8 (group0 128b, group1 256b; groups 2/3 unused
// for 2D tiles).
// ---------------------------------------------------------------------------
__device__ __forceinline__ void tdm_load_tile_f32(
    unsigned int lds_off, const float* gptr,
    unsigned int tile_x, unsigned int tile_y,
    unsigned int tensor_x, unsigned int tensor_y,
    unsigned int row_stride_elts) {
  const unsigned long long ga = (unsigned long long)(size_t)gptr;

  tdm_v4u g0 = {
      1u,                                                   // count=1 (valid D#)
      lds_off,                                              // lds_addr (bytes)
      (unsigned int)ga,                                     // global_addr[31:0]
      (unsigned int)((ga >> 32) & 0x1FFFFFFu) | (2u << 30)  // addr[56:32] | type=2
  };
  tdm_v8i g1 = {
      (int)(2u << 16),                                      // wg_mask=0, data_size=4B
      (int)((tensor_x & 0xFFFFu) << 16),                    // tensor_dim0[15:0]
      (int)(((tensor_x >> 16) & 0xFFFFu) |
            ((tensor_y & 0xFFFFu) << 16)),                  // dim0[31:16] | dim1[15:0]
      (int)(((tensor_y >> 16) & 0xFFFFu) |
            ((tile_x & 0xFFFFu) << 16)),                    // dim1[31:16] | tile_dim0
      (int)(tile_y & 0xFFFFu),                              // tile_dim1 (tile_dim2=0)
      (int)row_stride_elts,                                 // tensor_dim0_stride[31:0]
      0,                                                    // stride hi / dim1_stride lo
      0
  };
  tdm_v4i g2 = {0, 0, 0, 0};
  tdm_v4i g3 = {0, 0, 0, 0};
  tdm_v8i g4 = {0, 0, 0, 0, 0, 0, 0, 0};
  __builtin_amdgcn_tensor_load_to_lds(g0, g1, g2, g3, g4, /*cpol=*/0);
}

// ---------------------------------------------------------------------------
// Kernel 1: attn_h[B,ATT] = h[B,RNN] @ W[ATT,RNN]^T + bias[ATT]
// One wave32 per 16x16 output tile, v_wmma_f32_16x16x32_f16, K-loop of 32.
// ---------------------------------------------------------------------------
__global__ void __launch_bounds__(32)
attn_h_wmma_kernel(const float* __restrict__ h,
                   const float* __restrict__ W,
                   const float* __restrict__ bias,
                   float* __restrict__ attn_h) {
  const int tileN = blockIdx.x;            // 0..ATT/16-1
  const int tileM = blockIdx.y;            // 0..B/16-1
  const int lane  = threadIdx.x;           // 0..31
  const int half  = lane >> 4;
  const int m     = lane & 15;

  const float* __restrict__ arow = h + (size_t)(tileM * 16 + m) * RNN_;
  const float* __restrict__ brow = W + (size_t)(tileN * 16 + m) * RNN_;

  v8f c = {};
  for (int k0 = 0; k0 < RNN_; k0 += 32) {
    v16h a{}, b{};
    const float* __restrict__ ap = arow + k0 + 8 * half;
    const float* __restrict__ bp = brow + k0 + 8 * half;
#pragma unroll
    for (int j = 0; j < 8; ++j) {
      a[j]     = (_Float16)ap[j];
      a[j + 8] = (_Float16)ap[16 + j];
    }
#pragma unroll
    for (int j = 0; j < 8; ++j) {
      b[j]     = (_Float16)bp[j];
      b[j + 8] = (_Float16)bp[16 + j];
    }
    c = __builtin_amdgcn_wmma_f32_16x16x32_f16(
        /*neg_a=*/false, a, /*neg_b=*/false, b,
        /*c_mod=*/(short)0, c, /*reuse_a=*/false, /*reuse_b=*/false);
  }

  const float bn = bias[tileN * 16 + m];
#pragma unroll
  for (int r = 0; r < 8; ++r) {
    const int row = tileM * 16 + r + 8 * half;
    attn_h[(size_t)row * ATT_ + tileN * 16 + m] = c[r] + bn;
  }
}

// ---------------------------------------------------------------------------
// Kernel 2: scores[b,s] = w_alpha . tanh(proj[b,s,:] + attn_h[b,:]) + b_alpha
//           (masked with MIN_VALUE). One wave per s; 8 waves / block.
// Streams proj_context (268 MB) with b128 loads; attn_h/w_alpha in LDS.
// ---------------------------------------------------------------------------
#define SC_WAVES 8
__global__ void __launch_bounds__(256)
scores_kernel(const float* __restrict__ proj,
              const float* __restrict__ attn_h,
              const float* __restrict__ w_alpha,
              const float* __restrict__ b_alpha,
              const int*   __restrict__ mask,
              float* __restrict__ scores) {
  __shared__ float sh_ah[ATT_];
  __shared__ float sh_wa[ATT_];

  const int b   = blockIdx.y;
  const int tid = threadIdx.x;
  for (int i = tid; i < ATT_; i += 256) {
    sh_ah[i] = attn_h[(size_t)b * ATT_ + i];
    sh_wa[i] = w_alpha[i];
  }
  __syncthreads();

  const int wave = tid >> 5;
  const int lane = tid & 31;
  const int s    = blockIdx.x * SC_WAVES + wave;

  const float* __restrict__ p = proj + ((size_t)b * S_ + s) * ATT_;

  float acc = 0.0f;
#pragma unroll
  for (int v = 0; v < 4; ++v) {
    const int base = lane * 4 + v * 128;             // 2-way LDS bank pattern
    const float4 pv = *reinterpret_cast<const float4*>(p + base);
    acc += sh_wa[base + 0] * tanhf(pv.x + sh_ah[base + 0]);
    acc += sh_wa[base + 1] * tanhf(pv.y + sh_ah[base + 1]);
    acc += sh_wa[base + 2] * tanhf(pv.z + sh_ah[base + 2]);
    acc += sh_wa[base + 3] * tanhf(pv.w + sh_ah[base + 3]);
  }
#pragma unroll
  for (int off = 16; off > 0; off >>= 1) acc += __shfl_down(acc, off, 32);

  if (lane == 0) {
    float sc = acc + b_alpha[0];
    if (mask[(size_t)b * S_ + s]) sc = MIN_VALUE_;
    scores[(size_t)b * S_ + s] = sc;
  }
}

// ---------------------------------------------------------------------------
// Kernel 3: attn[b,:] = softmax(scores[b,:]) over S. One block per batch.
// ---------------------------------------------------------------------------
__global__ void __launch_bounds__(256)
softmax_kernel(const float* __restrict__ scores,
               float* __restrict__ attn) {
  __shared__ float red[8];
  const int b    = blockIdx.x;
  const int tid  = threadIdx.x;
  const int lane = tid & 31;
  const int wave = tid >> 5;

  float m = -3.4e38f;
  for (int s = tid; s < S_; s += 256) m = fmaxf(m, scores[(size_t)b * S_ + s]);
#pragma unroll
  for (int off = 16; off > 0; off >>= 1) m = fmaxf(m, __shfl_xor(m, off, 32));
  if (lane == 0) red[wave] = m;
  __syncthreads();
  float mb = red[0];
#pragma unroll
  for (int i = 1; i < 8; ++i) mb = fmaxf(mb, red[i]);
  __syncthreads();

  float sum = 0.0f;
  for (int s = tid; s < S_; s += 256) {
    const float e = __expf(scores[(size_t)b * S_ + s] - mb);
    attn[(size_t)b * S_ + s] = e;
    sum += e;
  }
#pragma unroll
  for (int off = 16; off > 0; off >>= 1) sum += __shfl_xor(sum, off, 32);
  if (lane == 0) red[wave] = sum;
  __syncthreads();
  float tot = 0.0f;
#pragma unroll
  for (int i = 0; i < 8; ++i) tot += red[i];
  const float inv = 1.0f / tot;
  for (int s = tid; s < S_; s += 256) attn[(size_t)b * S_ + s] *= inv;
}

// ---------------------------------------------------------------------------
// Kernel 4: wc[b,d] = sum_s attn[b,s] * context[b,s,d]
// One block per (b, 128-wide d chunk). context streamed by the Tensor Data
// Mover into double-buffered 32x128 f32 LDS tiles (wave 0 issues the DMA,
// s_wait_tensorcnt pipelines it against the FMA loop). LDS reads are
// conflict-free (consecutive lanes -> consecutive banks).
// ---------------------------------------------------------------------------
#define WS_CHUNK 32
#define WS_NCHUNK (S_ / WS_CHUNK)
__global__ void __launch_bounds__(128)
wctx_kernel(const float* __restrict__ attn,
            const float* __restrict__ ctx,
            float* __restrict__ out) {
  __shared__ float tile[2][WS_CHUNK * 128];
  __shared__ float sh_a[WS_CHUNK];

  const int b   = blockIdx.y;
  const int d0  = blockIdx.x * 128;
  const int tid = threadIdx.x;

  // Prologue: DMA chunk 0 into buffer 0.
  if (tid < 32) {
    tdm_load_tile_f32((unsigned int)(size_t)&tile[0][0],
                      ctx + ((size_t)b * S_ + 0) * CTX_ + d0,
                      /*tile_x=*/128, /*tile_y=*/WS_CHUNK,
                      /*tensor_x=*/CTX_, /*tensor_y=*/S_,
                      /*row_stride=*/CTX_);
  }

  float acc = 0.0f;
  for (int i = 0; i < WS_NCHUNK; ++i) {
    const int buf = i & 1;
    if (tid < 32) {
      if (i + 1 < WS_NCHUNK) {
        // Issue next chunk into the other buffer, then wait for the current
        // one (<=1 outstanding): double-buffered DMA pipeline.
        tdm_load_tile_f32((unsigned int)(size_t)&tile[buf ^ 1][0],
                          ctx + ((size_t)b * S_ + (i + 1) * WS_CHUNK) * CTX_ + d0,
                          128, WS_CHUNK, CTX_, S_, CTX_);
        __builtin_amdgcn_s_wait_tensorcnt(1);
      } else {
        __builtin_amdgcn_s_wait_tensorcnt(0);
      }
      sh_a[tid] = attn[(size_t)b * S_ + i * WS_CHUNK + tid];  // WS_CHUNK == 32
    }
    __syncthreads();

#pragma unroll 8
    for (int j = 0; j < WS_CHUNK; ++j) {
      acc = fmaf(sh_a[j], tile[buf][j * 128 + tid], acc);
    }
    __syncthreads();
  }
  out[(size_t)b * CTX_ + d0 + tid] = acc;
}

// ---------------------------------------------------------------------------
// Launch
// ---------------------------------------------------------------------------
extern "C" void kernel_launch(void* const* d_in, const int* in_sizes, int n_in,
                              void* d_out, int out_size, void* d_ws, size_t ws_size,
                              hipStream_t stream) {
  const float* h       = (const float*)d_in[0];   // [B, RNN]
  const float* proj    = (const float*)d_in[1];   // [B, S, ATT]
  const float* ctx     = (const float*)d_in[2];   // [B, S, CTX]
  const int*   mask    = (const int*)d_in[3];     // [B, S]
  const float* W       = (const float*)d_in[4];   // [ATT, RNN]
  const float* bias    = (const float*)d_in[5];   // [ATT]
  const float* w_alpha = (const float*)d_in[6];   // [ATT]
  const float* b_alpha = (const float*)d_in[7];   // scalar

  float* wc_out   = (float*)d_out;                       // [B, CTX]
  float* attn_out = (float*)d_out + (size_t)B_ * CTX_;   // [B, S]

  float* ws_attn_h = (float*)d_ws;                           // [B, ATT]
  float* ws_scores = (float*)d_ws + (size_t)B_ * ATT_;       // [B, S]

  // K1: WMMA GEMM for attn_h
  attn_h_wmma_kernel<<<dim3(ATT_ / 16, B_ / 16), 32, 0, stream>>>(
      h, W, bias, ws_attn_h);

  // K2: masked additive-attention scores (streams proj_context)
  scores_kernel<<<dim3(S_ / SC_WAVES, B_), 256, 0, stream>>>(
      proj, ws_attn_h, w_alpha, b_alpha, mask, ws_scores);

  // K3: per-batch softmax -> attn output slot
  softmax_kernel<<<dim3(B_), 256, 0, stream>>>(ws_scores, attn_out);

  // K4: weighted context (TDM-streamed context tiles)
  wctx_kernel<<<dim3(CTX_ / 128, B_), 128, 0, stream>>>(attn_out, ctx, wc_out);
}
